// MultiHeadDuplexAttention1_1219770712285
// MI455X (gfx1250) — matrix operand order repaired and checked by reading, not compiled, not run on hardware
//
#include <hip/hip_runtime.h>

// ---------------------------------------------------------------------------
// MultiHeadDuplexAttention — CDNA5 (gfx1250).
// Matmuls on v_wmma_f32_16x16x32_f16 (f16 in, f32 accumulate).
// All GEMMs are "NT" (A[M,K] row-major x Bt[N,K] row-major) so every WMMA
// fragment is contiguous along K. Tiles are staged global->LDS with the
// Tensor Data Mover (tensor_load_to_lds, TENSORcnt, double-buffered), with
// descriptor padding so LDS fragment reads are bank-conflict-free.
// ---------------------------------------------------------------------------

typedef __attribute__((ext_vector_type(16))) _Float16 v16h;
typedef __attribute__((ext_vector_type(8)))  _Float16 v8h;
typedef __attribute__((ext_vector_type(8)))  float    v8f;
typedef unsigned int v4u  __attribute__((ext_vector_type(4)));
typedef int          v8i_ __attribute__((ext_vector_type(8)));
typedef int          v4i_ __attribute__((ext_vector_type(4)));

#define DIM   1024
#define HEADS 16
#define DK    64
#define BATCH 4
#define SEQ   1024
#define MTOT  (BATCH*SEQ)          // 4096
#define SCALE 0.125f               // 1/sqrt(64)
#define EPS_LN 1e-5f
#define EPS_OM 1e-5f

// ---------------------------------------------------------------------------
// Tensor Data Mover: 2D tile (tile_k contiguous f16 x tile_rows rows) -> LDS.
// pad_iv/pad_amt: descriptor LDS padding codes (pad_iv: 3 => every 16 DWORDs,
// 4 => every 32 DWORDs; pad_amt: 3 => 4 DWORDs) -> padded LDS row strides.
// ---------------------------------------------------------------------------
__device__ __forceinline__ unsigned lds_off(const void* p) {
  return (unsigned)(unsigned long long)p;
}

__device__ __forceinline__ void tdm_load_2d(unsigned lds_addr, const void* gsrc,
                                            unsigned tile_k, unsigned tile_rows,
                                            unsigned long long row_stride_elems,
                                            unsigned pad_iv, unsigned pad_amt) {
  unsigned long long ga = (unsigned long long)gsrc;
  const unsigned td0 = 1u << 20, td1 = 1u << 20;   // generous bounds (no clip)
  v4u g0;
  g0[0] = 1u;                                       // count=1, user mode
  g0[1] = lds_addr;
  g0[2] = (unsigned)(ga & 0xFFFFFFFFu);
  g0[3] = (unsigned)((ga >> 32) & 0x01FFFFFFu) | 0x80000000u;  // type=2
  v8i_ g1;
  g1[0] = (int)(0x00010000u |                       // data_size = 2 bytes
                (1u << 20) | (pad_iv << 22) | (pad_amt << 25)); // pad_enable
  g1[1] = (int)((td0 & 0xFFFFu) << 16);
  g1[2] = (int)(((td0 >> 16) & 0xFFFFu) | ((td1 & 0xFFFFu) << 16));
  g1[3] = (int)(((td1 >> 16) & 0xFFFFu) | (tile_k << 16));
  g1[4] = (int)tile_rows;                           // tile_dim1 (tile_dim2=0)
  g1[5] = (int)(row_stride_elems & 0xFFFFFFFFull);  // tensor_dim0_stride lo
  g1[6] = (int)((row_stride_elems >> 32) & 0xFFFFull);
  g1[7] = 0;
  v4i_ z4 = {0, 0, 0, 0};
  v8i_ z8 = {0, 0, 0, 0, 0, 0, 0, 0};
  __builtin_amdgcn_tensor_load_to_lds(g0, g1, z4, z4, z8, 0);
}

// ---------------------------------------------------------------------------
// WMMA fragment helpers (layouts per CDNA5 ISA 7.12.2, wave32)
// ---------------------------------------------------------------------------
__device__ __forceinline__ v16h cat8(v8h lo, v8h hi) {
  v16h r;
#pragma unroll
  for (int i = 0; i < 8; ++i) { r[i] = lo[i]; r[i + 8] = hi[i]; }
  return r;
}
// A-frag: base = &row[g*8]; halves {base..+7, base+16..+23}
__device__ __forceinline__ v16h lds_afrag(const _Float16* base) {
  return cat8(*(const v8h*)base, *(const v8h*)(base + 16));
}
// B-frag: base = &row[g*8]; halves {base..+7, base+16..+23}
__device__ __forceinline__ v16h lds_bfrag(const _Float16* base) {
  return cat8(*(const v8h*)base, *(const v8h*)(base + 16));
}
// A-frag straight from global (used for Q)
__device__ __forceinline__ v16h glob_afrag(const _Float16* __restrict__ A,
                                           int lda, int row, int kb, int lane) {
  const _Float16* base = A + (size_t)row * lda + kb + ((lane >> 4) & 1) * 8;
  return cat8(*(const v8h*)base, *(const v8h*)(base + 16));
}

__device__ __forceinline__ v8f wmma_f16(v16h a, v16h b, v8f c) {
  return __builtin_amdgcn_wmma_f32_16x16x32_f16(false, a, false, b,
                                                (short)0, c, false, false);
}

__device__ __forceinline__ float rowred_max16(float x) {
#pragma unroll
  for (int m = 1; m < 16; m <<= 1) x = fmaxf(x, __shfl_xor(x, m, 32));
  return x;
}

// ---------------------------------------------------------------------------
// Batched NT GEMM: C[M,N] = A[M,K] * Bt[N,K]^T + bias
// block 256 thr = 8 waves; wave computes 32(M) x 64(N); block tile 256x64.
// A/B tiles TDM-staged into LDS, double-buffered; 8 WMMA per 32-deep K step.
// MODE: 0 = f32 out, 1 = f16 out, 2 = f16 transposed out (C^T, ldc=M)
// ---------------------------------------------------------------------------
#define GR 256   // block rows
#define PA 40    // padded LDS stride for 32-halve rows
#define PK 72    // padded LDS stride for 64-halve rows

template <int MODE>
__global__ __launch_bounds__(256) void k_gemm_nt(
    const _Float16* __restrict__ A, int lda, long long sA,
    const _Float16* __restrict__ Bt, int ldb, long long sB,
    const float* __restrict__ bias,
    void* __restrict__ Cout, int ldc, long long sC, int K) {
  __shared__ _Float16 As[2][GR][PA];
  __shared__ _Float16 Bs[2][64][PA];
  int lane = threadIdx.x & 31, wave = threadIdx.x >> 5;
  int g = lane >> 4;
  int bz = blockIdx.z;
  const _Float16* Ab = A + sA * bz + (size_t)blockIdx.y * GR * lda;
  const _Float16* Bb = Bt + sB * bz + (size_t)blockIdx.x * 64 * ldb;
  int row0 = blockIdx.y * GR;
  int col0 = blockIdx.x * 64;
  bool issuer = (wave == 0);

  v8f acc[2][4];
#pragma unroll
  for (int s = 0; s < 2; ++s)
#pragma unroll
    for (int j = 0; j < 4; ++j)
#pragma unroll
      for (int v = 0; v < 8; ++v) acc[s][j][v] = 0.f;

  if (issuer) {
    tdm_load_2d(lds_off(&As[0][0][0]), Ab, 32, GR, (unsigned long long)lda, 3, 3);
    tdm_load_2d(lds_off(&Bs[0][0][0]), Bb, 32, 64, (unsigned long long)ldb, 3, 3);
  }
  int cur = 0;
  for (int kb = 0; kb < K; kb += 32) {
    if (issuer) {
      if (kb + 32 < K) {
        tdm_load_2d(lds_off(&As[cur ^ 1][0][0]), Ab + kb + 32, 32, GR,
                    (unsigned long long)lda, 3, 3);
        tdm_load_2d(lds_off(&Bs[cur ^ 1][0][0]), Bb + kb + 32, 32, 64,
                    (unsigned long long)ldb, 3, 3);
        __builtin_amdgcn_s_wait_tensorcnt((short)2);
      } else {
        __builtin_amdgcn_s_wait_tensorcnt((short)0);
      }
    }
    __syncthreads();
    v16h af[2];
#pragma unroll
    for (int s = 0; s < 2; ++s)
      af[s] = lds_afrag(&As[cur][wave * 32 + s * 16 + (lane & 15)][g * 8]);
#pragma unroll
    for (int j = 0; j < 4; ++j) {
      v16h bf = lds_bfrag(&Bs[cur][j * 16 + (lane & 15)][g * 8]);
      acc[0][j] = wmma_f16(af[0], bf, acc[0][j]);
      acc[1][j] = wmma_f16(af[1], bf, acc[1][j]);
    }
    __syncthreads();
    cur ^= 1;
  }

#pragma unroll
  for (int s = 0; s < 2; ++s)
#pragma unroll
    for (int j = 0; j < 4; ++j) {
      int col = col0 + j * 16 + (lane & 15);
      float bv = bias ? bias[col] : 0.f;
#pragma unroll
      for (int v = 0; v < 8; ++v) {
        int row = row0 + wave * 32 + s * 16 + v + 8 * g;
        float val = acc[s][j][v] + bv;
        if (MODE == 0)
          ((float*)Cout)[sC * bz + (size_t)row * ldc + col] = val;
        else if (MODE == 1)
          ((_Float16*)Cout)[sC * bz + (size_t)row * ldc + col] = (_Float16)val;
        else
          ((_Float16*)Cout)[sC * bz + (size_t)col * ldc + row] = (_Float16)val;
      }
    }
}

// ---------------------------------------------------------------------------
// Fused per-head flash attention: O = softmax(Q Kh^T * SCALE) Vh
// Q, Km: [MTOT, DIM] f16, head = 64 cols at h*64. Vt: [DIM, MTOT] f16.
// K/V tiles are shared by all 8 waves -> TDM-staged once per 32-key step.
// Softmax denominator accumulated by WMMA against an all-ones B fragment.
// grid = (SEQ/128, HEADS, BATCH); wave = one 16-row q tile.
// ---------------------------------------------------------------------------
__global__ __launch_bounds__(256) void k_flash_head(
    const _Float16* __restrict__ Q, const _Float16* __restrict__ Km,
    const _Float16* __restrict__ Vt, _Float16* __restrict__ O) {
  __shared__ _Float16 Kt[2][32][PK];    // 32 keys x 64 head dims (padded)
  __shared__ _Float16 Vs[2][64][PA];    // 64 head dims x 32 keys (padded)
  __shared__ _Float16 pbuf[8][16][PA];  // per-wave P tile 16x32 (padded)
  int lane = threadIdx.x & 31, wave = threadIdx.x >> 5;
  int g = lane >> 4;
  int h = blockIdx.y, b = blockIdx.z;
  int hc = h * DK;
  int q0 = b * SEQ + blockIdx.x * 128 + wave * 16;
  bool issuer = (wave == 0);

  v16h qf0 = glob_afrag(Q, DIM, q0 + (lane & 15), hc + 0, lane);
  v16h qf1 = glob_afrag(Q, DIM, q0 + (lane & 15), hc + 32, lane);
  v16h ones;
#pragma unroll
  for (int i = 0; i < 16; ++i) ones[i] = (_Float16)1.0f;

  v8f acc[4], accl;
  float mi[8];
#pragma unroll
  for (int v = 0; v < 8; ++v) { mi[v] = -1e30f; accl[v] = 0.f; }
#pragma unroll
  for (int j = 0; j < 4; ++j)
#pragma unroll
    for (int v = 0; v < 8; ++v) acc[j][v] = 0.f;

  const _Float16* Kg = Km + (size_t)(b * SEQ) * DIM + hc;  // [keys][64]
  const _Float16* Vg = Vt + (size_t)hc * MTOT + b * SEQ;   // [64][keys]
  if (issuer) {
    tdm_load_2d(lds_off(&Kt[0][0][0]), Kg, 64, 32, (unsigned long long)DIM, 4, 3);
    tdm_load_2d(lds_off(&Vs[0][0][0]), Vg, 32, 64, (unsigned long long)MTOT, 3, 3);
  }
  int cur = 0;
  for (int kb = 0; kb < SEQ; kb += 32) {
    if (issuer) {
      if (kb + 32 < SEQ) {
        tdm_load_2d(lds_off(&Kt[cur ^ 1][0][0]), Kg + (size_t)(kb + 32) * DIM,
                    64, 32, (unsigned long long)DIM, 4, 3);
        tdm_load_2d(lds_off(&Vs[cur ^ 1][0][0]), Vg + kb + 32,
                    32, 64, (unsigned long long)MTOT, 3, 3);
        __builtin_amdgcn_s_wait_tensorcnt((short)2);
      } else {
        __builtin_amdgcn_s_wait_tensorcnt((short)0);
      }
    }
    __syncthreads();
    // ---- scores: two 16-key blocks, K-dim = 64 -> 2 chained WMMAs each
    v8f s[2];
#pragma unroll
    for (int j = 0; j < 2; ++j) {
      v16h kf0 = lds_bfrag(&Kt[cur][j * 16 + (lane & 15)][0 + g * 8]);
      v16h kf1 = lds_bfrag(&Kt[cur][j * 16 + (lane & 15)][32 + g * 8]);
      v8f z;
#pragma unroll
      for (int v = 0; v < 8; ++v) z[v] = 0.f;
      z = wmma_f16(qf0, kf0, z);
      z = wmma_f16(qf1, kf1, z);
      s[j] = z;
    }
    // ---- online softmax; row = v + 8g, its 16 cols live in 16 lanes
    float corr[8];
#pragma unroll
    for (int v = 0; v < 8; ++v) {
      float s0 = s[0][v] * SCALE, s1 = s[1][v] * SCALE;
      float mx = rowred_max16(fmaxf(s0, s1));
      float mn = fmaxf(mi[v], mx);
      corr[v] = __expf(mi[v] - mn);
      mi[v] = mn;
      float p0 = __expf(s0 - mn), p1 = __expf(s1 - mn);
      int prow = v + 8 * g;
      pbuf[wave][prow][lane & 15] = (_Float16)p0;
      pbuf[wave][prow][16 + (lane & 15)] = (_Float16)p1;
    }
#pragma unroll
    for (int j = 0; j < 4; ++j)
#pragma unroll
      for (int v = 0; v < 8; ++v) acc[j][v] *= corr[v];
#pragma unroll
    for (int v = 0; v < 8; ++v) accl[v] *= corr[v];
    __syncthreads();
    // ---- P as A-fragment (transposed through LDS), then P@V + P@1
    v16h pf = lds_afrag(&pbuf[wave][lane & 15][g * 8]);
#pragma unroll
    for (int j = 0; j < 4; ++j) {
      v16h vf = lds_bfrag(&Vs[cur][j * 16 + (lane & 15)][g * 8]);
      acc[j] = wmma_f16(pf, vf, acc[j]);
    }
    accl = wmma_f16(pf, ones, accl);   // softmax denominator, no shuffles
    __syncthreads();
    cur ^= 1;
  }

#pragma unroll
  for (int j = 0; j < 4; ++j)
#pragma unroll
    for (int v = 0; v < 8; ++v) {
      int row = q0 + v + 8 * g;
      int col = hc + j * 16 + (lane & 15);
      O[(size_t)row * DIM + col] = (_Float16)(acc[j][v] / accl[v]);
    }
}

// ---------------------------------------------------------------------------
// Elementwise / reduction kernels
// ---------------------------------------------------------------------------
__global__ __launch_bounds__(256) void k_cvt(const float* __restrict__ in,
                                             _Float16* __restrict__ out, int n) {
  int i = blockIdx.x * 256 + threadIdx.x;
  if (i < n) out[i] = (_Float16)in[i];
}

__global__ __launch_bounds__(256) void k_twT(const float* __restrict__ W,
                                             _Float16* __restrict__ Wt,
                                             int rows, int cols) {
  __shared__ float t[32][33];
  int c0 = blockIdx.x * 32, r0 = blockIdx.y * 32;
  int x = threadIdx.x & 31, y = threadIdx.x >> 5;
#pragma unroll
  for (int i = 0; i < 32; i += 8) t[y + i][x] = W[(size_t)(r0 + y + i) * cols + c0 + x];
  __syncthreads();
#pragma unroll
  for (int i = 0; i < 32; i += 8)
    Wt[(size_t)(c0 + y + i) * rows + r0 + x] = (_Float16)t[x][y + i];
}

__global__ __launch_bounds__(256) void k_tr16(const _Float16* __restrict__ in,
                                              _Float16* __restrict__ out,
                                              int rows, int cols) {
  __shared__ _Float16 t[32][33];
  int c0 = blockIdx.x * 32, r0 = blockIdx.y * 32;
  int x = threadIdx.x & 31, y = threadIdx.x >> 5;
#pragma unroll
  for (int i = 0; i < 32; i += 8) t[y + i][x] = in[(size_t)(r0 + y + i) * cols + c0 + x];
  __syncthreads();
#pragma unroll
  for (int i = 0; i < 32; i += 8) out[(size_t)(c0 + y + i) * rows + r0 + x] = t[x][y + i];
}

__global__ __launch_bounds__(256) void k_softmax_rows(const float* __restrict__ in,
                                                      _Float16* __restrict__ out,
                                                      int n) {
  __shared__ float red[256];
  int row = blockIdx.x, t = threadIdx.x;
  const float* x = in + (size_t)row * n;
  _Float16* y = out + (size_t)row * n;
  float mx = -1e30f;
  for (int i = t; i < n; i += 256) mx = fmaxf(mx, x[i] * SCALE);
  red[t] = mx; __syncthreads();
  for (int k = 128; k > 0; k >>= 1) { if (t < k) red[t] = fmaxf(red[t], red[t + k]); __syncthreads(); }
  mx = red[0]; __syncthreads();
  float sum = 0.f;
  for (int i = t; i < n; i += 256) sum += __expf(x[i] * SCALE - mx);
  red[t] = sum; __syncthreads();
  for (int k = 128; k > 0; k >>= 1) { if (t < k) red[t] += red[t + k]; __syncthreads(); }
  float inv = 1.f / red[0];
  for (int i = t; i < n; i += 256) y[i] = (_Float16)(__expf(x[i] * SCALE - mx) * inv);
}

__global__ __launch_bounds__(256) void k_ln_residual(const float* __restrict__ Yin,
                                                     const float* __restrict__ A,
                                                     float* __restrict__ o32,
                                                     _Float16* __restrict__ o16) {
  __shared__ float red[256];
  int row = blockIdx.x, t = threadIdx.x;
  const float* y = Yin + (size_t)row * DIM;
  const float* a = A + (size_t)row * DIM;
  float z[4], s = 0.f;
#pragma unroll
  for (int i = 0; i < 4; ++i) { z[i] = y[t + i * 256] + a[t + i * 256]; s += z[i]; }
  red[t] = s; __syncthreads();
  for (int k = 128; k > 0; k >>= 1) { if (t < k) red[t] += red[t + k]; __syncthreads(); }
  float mu = red[0] / DIM; __syncthreads();
  float ss = 0.f;
#pragma unroll
  for (int i = 0; i < 4; ++i) { float d = z[i] - mu; ss += d * d; }
  red[t] = ss; __syncthreads();
  for (int k = 128; k > 0; k >>= 1) { if (t < k) red[t] += red[t + k]; __syncthreads(); }
  float rstd = rsqrtf(red[0] / DIM + EPS_LN);
#pragma unroll
  for (int i = 0; i < 4; ++i) {
    float o = (z[i] - mu) * rstd;
    o32[(size_t)row * DIM + t + i * 256] = o;
    o16[(size_t)row * DIM + t + i * 256] = (_Float16)o;
  }
}

__global__ __launch_bounds__(256) void k_colstats(const float* __restrict__ X,
                                                  float* __restrict__ mu,
                                                  float* __restrict__ inv) {
  int d = blockIdx.x * 256 + threadIdx.x;
  int b = blockIdx.y;
  const float* x = X + (size_t)b * SEQ * DIM + d;
  float s = 0.f, ss = 0.f;
  for (int i = 0; i < SEQ; ++i) { float v = x[(size_t)i * DIM]; s += v; ss += v * v; }
  float m = s / SEQ;
  float var = (ss - (float)SEQ * m * m) / (float)(SEQ - 1);
  float sig = sqrtf(fmaxf(var, 0.f));
  mu[b * DIM + d] = m;
  inv[b * DIM + d] = 1.f / (sig + EPS_OM);
}

__global__ __launch_bounds__(256) void k_combine(const float* __restrict__ G,
                                                 const float* __restrict__ Bv,
                                                 const float* __restrict__ X,
                                                 const float* __restrict__ mu,
                                                 const float* __restrict__ inv,
                                                 float* __restrict__ out) {
  size_t i = (size_t)blockIdx.x * 256 + threadIdx.x;
  int d = (int)(i % DIM);
  int m = (int)(i / DIM);
  int b = m / SEQ;
  float om = (X[i] - mu[b * DIM + d]) * inv[b * DIM + d];
  out[i] = G[i] * om + Bv[i];
}

// ---------------------------------------------------------------------------
// Host orchestration
// ---------------------------------------------------------------------------
extern "C" void kernel_launch(void* const* d_in, const int* in_sizes, int n_in,
                              void* d_out, int out_size, void* d_ws, size_t ws_size,
                              hipStream_t stream) {
  (void)in_sizes; (void)n_in; (void)out_size; (void)ws_size;
  const float* X  = (const float*)d_in[0];
  const float* Y  = (const float*)d_in[1];
  const float* Wq = (const float*)d_in[2];  const float* bq = (const float*)d_in[3];
  const float* Wk = (const float*)d_in[4];  const float* bk = (const float*)d_in[5];
  const float* Wv = (const float*)d_in[6];  const float* bv = (const float*)d_in[7];
  const float* Wo = (const float*)d_in[8];  const float* bo = (const float*)d_in[9];
  const float* Wg = (const float*)d_in[10]; const float* bg = (const float*)d_in[11];
  const float* Wb = (const float*)d_in[12]; const float* bb = (const float*)d_in[13];
  float* outX = (float*)d_out;
  float* outY = (float*)d_out + (size_t)MTOT * DIM;

  const size_t MD  = (size_t)MTOT * DIM;
  const size_t DD  = (size_t)DIM * DIM;
  const size_t BSS = (size_t)BATCH * SEQ * SEQ;

  char* p = (char*)d_ws;
  auto alloc = [&](size_t bytes) -> void* {
    void* r = (void*)p;
    p += (bytes + 255) & ~(size_t)255;
    return r;
  };
  _Float16* WtQ = (_Float16*)alloc(DD * 2);
  _Float16* WtK = (_Float16*)alloc(DD * 2);
  _Float16* WtV = (_Float16*)alloc(DD * 2);
  _Float16* WtO = (_Float16*)alloc(DD * 2);
  _Float16* WtG = (_Float16*)alloc(DD * 2);
  _Float16* WtB = (_Float16*)alloc(DD * 2);
  _Float16* X16   = (_Float16*)alloc(MD * 2);
  _Float16* Y16   = (_Float16*)alloc(MD * 2);
  _Float16* Q1    = (_Float16*)alloc(MD * 2);
  _Float16* K1    = (_Float16*)alloc(MD * 2);   // X@Wk (shared by ua & compute_K)
  _Float16* Vt1   = (_Float16*)alloc(MD * 2);   // (X@Wv)^T  [D][M]
  _Float16* Att1  = (_Float16*)alloc(MD * 2);
  _Float16* Yn16  = (_Float16*)alloc(MD * 2);
  _Float16* YnT   = (_Float16*)alloc(MD * 2);   // Y_new^T [D][M]
  _Float16* Qc    = (_Float16*)alloc(MD * 2);
  _Float16* Q2    = (_Float16*)alloc(MD * 2);
  _Float16* P16   = (_Float16*)alloc(BSS * 2);
  _Float16* Kfull = (_Float16*)alloc(MD * 2);
  _Float16* Att2  = (_Float16*)alloc(MD * 2);
  _Float16* A2    = (_Float16*)alloc(MD * 2);
  float* buf32a = (float*)alloc(MD * 4);   // Aout(ua), then G
  float* buf32b = (float*)alloc(BSS * 4);  // scores, then Bv
  float* muC  = (float*)alloc((size_t)BATCH * DIM * 4);
  float* invC = (float*)alloc((size_t)BATCH * DIM * 4);

  dim3 blk(256);
  k_cvt<<<(unsigned)(MD / 256), blk, 0, stream>>>(X, X16, (int)MD);
  k_cvt<<<(unsigned)(MD / 256), blk, 0, stream>>>(Y, Y16, (int)MD);
  dim3 gW(DIM / 32, DIM / 32);
  k_twT<<<gW, blk, 0, stream>>>(Wq, WtQ, DIM, DIM);
  k_twT<<<gW, blk, 0, stream>>>(Wk, WtK, DIM, DIM);
  k_twT<<<gW, blk, 0, stream>>>(Wv, WtV, DIM, DIM);
  k_twT<<<gW, blk, 0, stream>>>(Wo, WtO, DIM, DIM);
  k_twT<<<gW, blk, 0, stream>>>(Wg, WtG, DIM, DIM);
  k_twT<<<gW, blk, 0, stream>>>(Wb, WtB, DIM, DIM);

  dim3 gP(DIM / 64, MTOT / GR, 1);       // projection GEMMs (M=4096)
  dim3 gF(SEQ / 128, HEADS, BATCH);      // flash attention

  // --- ua(Y, X): Y_new = LN(Y + Wo(attn(Y@Wq, X@Wk, X@Wv)) + bo)
  k_gemm_nt<1><<<gP, blk, 0, stream>>>(Y16, DIM, 0, WtQ, DIM, 0, bq, Q1, DIM, 0, DIM);
  k_gemm_nt<1><<<gP, blk, 0, stream>>>(X16, DIM, 0, WtK, DIM, 0, bk, K1, DIM, 0, DIM);
  k_gemm_nt<2><<<gP, blk, 0, stream>>>(X16, DIM, 0, WtV, DIM, 0, bv, Vt1, MTOT, 0, DIM);
  k_flash_head<<<gF, blk, 0, stream>>>(Q1, K1, Vt1, Att1);
  k_gemm_nt<0><<<gP, blk, 0, stream>>>(Att1, DIM, 0, WtO, DIM, 0, bo, buf32a, DIM, 0, DIM);
  k_ln_residual<<<MTOT, blk, 0, stream>>>(Y, buf32a, outY, Yn16);
  dim3 gT(DIM / 32, MTOT / 32);
  k_tr16<<<gT, blk, 0, stream>>>(Yn16, YnT, MTOT, DIM);

  // --- ud(X, Y_new)
  k_gemm_nt<1><<<gP, blk, 0, stream>>>(Yn16, DIM, 0, WtQ, DIM, 0, bq, Qc, DIM, 0, DIM);
  k_gemm_nt<1><<<gP, blk, 0, stream>>>(X16, DIM, 0, WtQ, DIM, 0, bq, Q2, DIM, 0, DIM);
  dim3 gS(SEQ / 64, SEQ / GR, BATCH);
  k_gemm_nt<0><<<gS, blk, 0, stream>>>(Qc, DIM, (long long)SEQ * DIM,
                                       K1, DIM, (long long)SEQ * DIM,
                                       nullptr, buf32b, SEQ, (long long)SEQ * SEQ, DIM);
  k_softmax_rows<<<BATCH * SEQ, blk, 0, stream>>>(buf32b, P16, SEQ);
  dim3 gK(DIM / 64, SEQ / GR, BATCH);
  k_gemm_nt<1><<<gK, blk, 0, stream>>>(P16, SEQ, (long long)SEQ * SEQ,
                                       YnT, MTOT, (long long)SEQ,
                                       nullptr, Kfull, DIM, (long long)SEQ * DIM, SEQ);
  k_flash_head<<<gF, blk, 0, stream>>>(Q2, Kfull, YnT, Att2);
  k_gemm_nt<1><<<gP, blk, 0, stream>>>(Att2, DIM, 0, WtO, DIM, 0, bo, A2, DIM, 0, DIM);
  k_gemm_nt<0><<<gP, blk, 0, stream>>>(A2, DIM, 0, WtG, DIM, 0, bg, buf32a, DIM, 0, DIM);
  k_gemm_nt<0><<<gP, blk, 0, stream>>>(A2, DIM, 0, WtB, DIM, 0, bb, buf32b, DIM, 0, DIM);
  dim3 gC(DIM / 256, BATCH);
  k_colstats<<<gC, blk, 0, stream>>>(X, muC, invC);
  k_combine<<<(unsigned)(MD / 256), blk, 0, stream>>>(buf32a, buf32b, X, muC, invC, outX);
}
